// Mix2Layer_33045478375559
// MI455X (gfx1250) — matrix operand order repaired and checked by reading, not compiled, not run on hardware
//
#include <hip/hip_runtime.h>
#include <hip/hip_bf16.h>

// CDNA5 / gfx1250 fused Mix2Layer:
//   g = relu(x @ w1 + b1)  [b,o,k];  out = sum_{i,k} x*w2*g + b2  [b,o]
// bf16 WMMA 16x16x32, f32 accumulation. Block = 8 waves = 128 b-rows x 8 o-cols.
// Pipelined: register-staged global loads + LDS double buffer, 4-deep rotating
// B-operand pipeline enforced with sched_group_barrier, final chunk peeled.

typedef __attribute__((ext_vector_type(16))) __bf16 v16bf;
typedef __attribute__((ext_vector_type(8)))  __bf16 v8bf;
typedef __attribute__((ext_vector_type(8)))  float  v8f;

constexpr int NB   = 2048;   // batch
constexpr int DIN  = 2048;
constexpr int DOUT = 2048;
constexpr int KG   = 16;     // gating dim
constexpr int OB   = 8;      // o columns per block
constexpr int BM   = 128;    // b rows per block (8 waves x 16)
constexpr int IC   = 32;     // i chunk = WMMA K
constexpr int PAD  = 40;     // padded inner dim: keeps all gathers 16B aligned
constexpr int NCH  = DIN / IC;
constexpr int WSTR = DOUT * KG;   // w i-stride in elements

__device__ __forceinline__ v16bf ld_pair(const __bf16* p0, const __bf16* p1) {
  v8bf lo = *(const v8bf*)p0;   // ds_load_b128
  v8bf hi = *(const v8bf*)p1;   // ds_load_b128
  return __builtin_shufflevector(lo, hi, 0,1,2,3,4,5,6,7,8,9,10,11,12,13,14,15);
}

__device__ __forceinline__ unsigned pack2_bf16(float a, float b) {
  union { __bf16 h[2]; unsigned u; } un;
  un.h[0] = (__bf16)a;
  un.h[1] = (__bf16)b;
  return un.u;
}

__global__ __launch_bounds__(256) void mix2_fused_kernel(
    const float* __restrict__ xg,  const float* __restrict__ w1g,
    const float* __restrict__ b1g, const float* __restrict__ w2g,
    const float* __restrict__ b2g, float* __restrict__ outg) {
  const int bbase = blockIdx.x * BM;   // b fastest -> 16 blocks share one o-group in L2
  const int obase = blockIdx.y * OB;
  const int t    = threadIdx.x;
  const int lane = t & 31;
  const int wave = t >> 5;             // wave owns b-subtile [wave*16, wave*16+16)
  const int m    = lane & 15;          // A row / B column (= gating k) / D N-index
  const int h    = lane >> 4;          // half-wave selector

  __shared__ __bf16 sA [2][BM][PAD];       // x tile   [b][i]
  __shared__ __bf16 sB1[2][OB][KG][PAD];   // w1 tile  [o][k][i] (transposed)
  __shared__ __bf16 sB2[2][OB][KG][PAD];   // w2 tile

  v8f c1[OB] = {};   // accum for x@w1 (pre-relu gate)
  v8f c2[OB] = {};   // accum for x@w2

  // ---- staging decomposition (256 threads) ----
  const int xrow = t >> 2;             // x: rows {xrow, xrow+64}
  const int ioct = t & 3;              // x: i = ioct*8 (8 consecutive)
  const int ok   = t & 127;            // w: one (o,k) column
  const int oo   = ok >> 4;
  const int kk   = ok & 15;
  const int wi0  = (t >> 7) * 4;       // w: i group base 0 or 4
  // running pointers, stepped by constants -> one 64-bit add per chunk
  const float* xp  = xg  + (bbase + xrow) * DIN + ioct * 8;
  const float* w1p = w1g + wi0 * WSTR + (obase + oo) * KG + kk;
  const float* w2p = w2g + wi0 * WSTR + (obase + oo) * KG + kk;

  float4 xa[2], xb[2];                 // x stage: 16 floats
  float  w1v[4][4], w2v[4][4];         // w stage: 32 floats

  auto load_stage = [&]() {
    #pragma unroll
    for (int s = 0; s < 2; ++s) {      // rows xrow + s*64
      xa[s] = *(const float4*)(xp + s * (64 * DIN));
      xb[s] = *(const float4*)(xp + s * (64 * DIN) + 4);
    }
    #pragma unroll
    for (int s = 0; s < 4; ++s) {      // i = wi0 + s*8 + j; row deltas fit IOFFSET
      #pragma unroll
      for (int j = 0; j < 4; ++j) {
        w1v[s][j] = w1p[(s * 8 + j) * WSTR];
        w2v[s][j] = w2p[(s * 8 + j) * WSTR];
      }
    }
  };

  auto store_stage = [&](int q) {
    #pragma unroll
    for (int s = 0; s < 2; ++s) {      // one ds_store_b128 per row-group
      uint4 u = make_uint4(pack2_bf16(xa[s].x, xa[s].y), pack2_bf16(xa[s].z, xa[s].w),
                           pack2_bf16(xb[s].x, xb[s].y), pack2_bf16(xb[s].z, xb[s].w));
      *(uint4*)&sA[q][xrow + s * 64][ioct * 8] = u;
    }
    #pragma unroll
    for (int s = 0; s < 4; ++s) {      // one ds_store_b64 per 4-row group, per tensor
      uint2 u1 = make_uint2(pack2_bf16(w1v[s][0], w1v[s][1]), pack2_bf16(w1v[s][2], w1v[s][3]));
      uint2 u2 = make_uint2(pack2_bf16(w2v[s][0], w2v[s][1]), pack2_bf16(w2v[s][2], w2v[s][3]));
      *(uint2*)&sB1[q][oo][kk][wi0 + s * 8] = u1;
      *(uint2*)&sB2[q][oo][kk][wi0 + s * 8] = u2;
    }
  };

  // 16 WMMAs per chunk ordered (B1 o=0, B2 o=0, B1 o=1, ...) with a 4-deep
  // rotating operand buffer (4 operands live -> ds loads stay ~3 pairs ahead).
  auto compute = [&](int q) {
    const __bf16* ap = &sA[q][wave * 16 + m][h * 16];
    v16bf amat = ld_pair(ap, ap + 8);    // A: x[m, h*16..+15]
    const __bf16* b1p = &sB1[q][0][m][h * 16];
    const __bf16* b2p = &sB2[q][0][m][h * 16];

    v16bf R[4];
    #pragma unroll
    for (int idx = 0; idx < 4; ++idx) {  // prime the pipeline
      const __bf16* np = ((idx & 1) ? b2p : b1p) + (idx >> 1) * (KG * PAD);
      R[idx] = ld_pair(np, np + 8);
    }
    #pragma unroll
    for (int idx = 0; idx < 16; ++idx) {
      const int r = idx & 3;
      const int o = idx >> 1;
      if (idx & 1)
        c2[o] = __builtin_amdgcn_wmma_f32_16x16x32_bf16(
            false, amat, false, R[r], (short)0, c2[o], false, false);
      else
        c1[o] = __builtin_amdgcn_wmma_f32_16x16x32_bf16(
            false, amat, false, R[r], (short)0, c1[o], false, false);
      if (idx + 4 < 16) {                // refill the slot just consumed
        const int nidx = idx + 4;
        const __bf16* np = ((nidx & 1) ? b2p : b1p) + (nidx >> 1) * (KG * PAD);
        R[r] = ld_pair(np, np + 8);
      }
    }
  };

  // ---- prologue: stage chunk 0 into buffer 0 ----
  load_stage();
  store_stage(0);
  __syncthreads();
  xp  += IC;
  w1p += IC * WSTR;
  w2p += IC * WSTR;

  // ---- steady state: load chunk n+1 | compute chunk n | store chunk n+1 ----
  int p = 0;
  for (int it = 0; it < NCH - 1; ++it) {
    load_stage();
    compute(p);
    store_stage(p ^ 1);

    // Pin the schedule for this region (counts match exactly):
    //   36 vmem reads | 10 ds reads (A + 4 primed B) |
    //   12 x (1 wmma, 2 ds reads) | 4 wmma | 10 ds writes
    __builtin_amdgcn_sched_group_barrier(0x020, 36, 0);  // VMEM read
    __builtin_amdgcn_sched_group_barrier(0x100, 10, 0);  // DS read
    #pragma unroll
    for (int u = 0; u < 12; ++u) {
      __builtin_amdgcn_sched_group_barrier(0x008, 1, 0); // WMMA
      __builtin_amdgcn_sched_group_barrier(0x100, 2, 0); // DS read
    }
    __builtin_amdgcn_sched_group_barrier(0x008, 4, 0);   // WMMA tail
    __builtin_amdgcn_sched_group_barrier(0x200, 10, 0);  // DS write

    __syncthreads();
    xp  += IC;
    w1p += IC * WSTR;
    w2p += IC * WSTR;
    p ^= 1;
  }
  compute(p);   // peeled final chunk: no staging, no barrier

  // ---- epilogue: relu gate, product, reduce over k (16 lanes of each half) ----
  // C/D layout: VGPR j -> row M = j + h*8 ; lane%16 -> N (= gating k).
  #pragma unroll
  for (int o = 0; o < OB; ++o) {
    const int   og    = obase + o;
    const float bias1 = b1g[og * KG + m];
    const float bias2 = b2g[og];
    #pragma unroll
    for (int j = 0; j < 8; ++j) {
      float gv = c1[o][j] + bias1;
      gv = gv > 0.f ? gv : 0.f;          // relu gate
      float pd = c2[o][j] * gv;
      pd += __shfl_xor(pd, 1, 32);       // masks <16: butterfly stays inside
      pd += __shfl_xor(pd, 2, 32);       // each 16-lane half (k-reduction)
      pd += __shfl_xor(pd, 4, 32);
      pd += __shfl_xor(pd, 8, 32);
      if (m == 0) {                      // lane 0 -> rows j ; lane 16 -> rows j+8
        const int row = bbase + wave * 16 + h * 8 + j;
        outg[row * DOUT + og] = pd + bias2;
      }
    }
  }
}

extern "C" void kernel_launch(void* const* d_in, const int* in_sizes, int n_in,
                              void* d_out, int out_size, void* d_ws, size_t ws_size,
                              hipStream_t stream) {
  const float* x  = (const float*)d_in[0];
  const float* w1 = (const float*)d_in[1];
  const float* b1 = (const float*)d_in[2];
  const float* w2 = (const float*)d_in[3];
  const float* b2 = (const float*)d_in[4];
  float* out = (float*)d_out;

  dim3 grid(NB / BM, DOUT / OB);   // (16, 256): b fastest for L2 reuse of w tiles
  dim3 block(256);                 // 8 wave32
  mix2_fused_kernel<<<grid, block, 0, stream>>>(x, w1, b1, w2, b2, out);
}
